// ChakraMaskedAttention_77275051589880
// MI455X (gfx1250) — matrix operand order repaired and checked by reading, not compiled, not run on hardware
//
#include <hip/hip_runtime.h>
#include <hip/hip_bf16.h>

// Problem constants (fixed by the reference).
#define BATCH 4
#define TSEQ  2048
#define CDIM  1024
#define NH    16
#define DH    64
#define MROWS (BATCH * TSEQ)   // 8192

typedef unsigned short u16;
typedef __attribute__((ext_vector_type(16))) __bf16 v16bf;
typedef __attribute__((ext_vector_type(8)))  __bf16 v8bf;
typedef __attribute__((ext_vector_type(8)))  float  v8f;
typedef int i32x4 __attribute__((vector_size(16)));

__device__ __forceinline__ u16 f2bf(float f) {
    unsigned int u = __float_as_uint(f);
    unsigned int r = u + 0x7FFFu + ((u >> 16) & 1u);   // round-to-nearest-even
    return (u16)(r >> 16);
}

__device__ __forceinline__ v16bf cat16(v8bf lo, v8bf hi) {
    return __builtin_shufflevector(lo, hi, 0, 1, 2, 3, 4, 5, 6, 7,
                                           8, 9, 10, 11, 12, 13, 14, 15);
}

__device__ __forceinline__ v8f wmma_bf16(v16bf a, v16bf b, v8f c) {
    // (neg_a, A, neg_b, B, c_mod, C, reuse_a, reuse_b)
    return __builtin_amdgcn_wmma_f32_16x16x32_bf16(false, a, false, b,
                                                   (short)0, c, false, false);
}

// ---- async global->LDS staging (CDNA5 ASYNCcnt path, with fallback) -------
__device__ __forceinline__ void stage16(void* lds_dst, const void* gsrc) {
#if __has_builtin(__builtin_amdgcn_global_load_async_to_lds_b128)
    __builtin_amdgcn_global_load_async_to_lds_b128(
        (__attribute__((address_space(1))) i32x4*)gsrc,
        (__attribute__((address_space(3))) i32x4*)lds_dst,
        0, 0);
#else
    *(v8bf*)lds_dst = *(const v8bf*)gsrc;   // fallback: vmem load + ds store
#endif
}

__device__ __forceinline__ void stage_fence() {
#if __has_builtin(__builtin_amdgcn_global_load_async_to_lds_b128)
#if __has_builtin(__builtin_amdgcn_s_wait_asynccnt)
    __builtin_amdgcn_s_wait_asynccnt(0);
#else
    asm volatile("s_wait_asynccnt 0" ::: "memory");
#endif
#endif
    __syncthreads();
}

// ---------------------------------------------------------------------------
// fp32 -> bf16 conversion (grid-stride)
// ---------------------------------------------------------------------------
__global__ void cvt_f32_bf16_kernel(const float* __restrict__ in,
                                    u16* __restrict__ out, int n) {
    int i = blockIdx.x * blockDim.x + threadIdx.x;
    int stride = gridDim.x * blockDim.x;
    for (; i < n; i += stride) out[i] = f2bf(in[i]);
}

// ---------------------------------------------------------------------------
// Y = X (MROWS x CDIM, bf16 rowmajor) @ W^T  (W is [CDIM out, CDIM in] bf16)
// One wave computes a 32x64 tile: 2 A row-tiles share 4 B column-tiles
// -> 8 v_wmma per K-step of 32.
// MODE 0: store f32 row-major [MROWS x CDIM]
// MODE 1: store bf16 to [B,H,T,HD]   (Q / K)
// MODE 2: store bf16 to [B,H,HD,T]   (V transposed)
// ---------------------------------------------------------------------------
template <int MODE>
__global__ __launch_bounds__(128)
void gemm_xwT_kernel(const u16* __restrict__ X, const u16* __restrict__ W,
                     float* __restrict__ Yf, u16* __restrict__ Yb) {
    const int wid  = threadIdx.x >> 5;
    const int lane = threadIdx.x & 31;
    const int half = lane >> 4;
    const int lr   = lane & 15;

    const int row0 = blockIdx.x * 32;                 // M tile (2 x 16)
    const int col0 = blockIdx.y * 256 + wid * 64;     // N tile (64 per wave)

    const u16* xrow0 = X + (size_t)(row0 + lr) * CDIM;
    const u16* xrow1 = X + (size_t)(row0 + 16 + lr) * CDIM;
    const u16* wrow[4];
#pragma unroll
    for (int nt = 0; nt < 4; ++nt)
        wrow[nt] = W + (size_t)(col0 + nt * 16 + lr) * CDIM;

    v8f acc[2][4] = {};

#pragma unroll 2
    for (int k0 = 0; k0 < CDIM; k0 += 32) {
        // A operands: row lr (+16), K chunks {0..7,16..23}/{8..15,24..31}
        v16bf a0 = cat16(*(const v8bf*)(xrow0 + k0 + half * 8),
                         *(const v8bf*)(xrow0 + k0 + 16 + half * 8));
        v16bf a1 = cat16(*(const v8bf*)(xrow1 + k0 + half * 8),
                         *(const v8bf*)(xrow1 + k0 + 16 + half * 8));
#pragma unroll
        for (int nt = 0; nt < 4; ++nt) {
            // B operand: column lr, K chunks {0..7,16..23}/{8..15,24..31}
            v16bf b = cat16(*(const v8bf*)(wrow[nt] + k0 + half * 8),
                            *(const v8bf*)(wrow[nt] + k0 + 16 + half * 8));
            acc[0][nt] = wmma_bf16(a0, b, acc[0][nt]);
            acc[1][nt] = wmma_bf16(a1, b, acc[1][nt]);
        }
    }

#pragma unroll
    for (int rt = 0; rt < 2; ++rt) {
#pragma unroll
        for (int r = 0; r < 8; ++r) {
            const int m = row0 + rt * 16 + r + half * 8;   // output row
            if (MODE == 0) {
                float* yr = Yf + (size_t)m * CDIM + col0;
#pragma unroll
                for (int nt = 0; nt < 4; ++nt) yr[nt * 16 + lr] = acc[rt][nt][r];
            } else {
                const int t = m & (TSEQ - 1);
                const int b = m >> 11;                     // TSEQ = 2048
#pragma unroll
                for (int nt = 0; nt < 4; ++nt) {
                    const int n = col0 + nt * 16 + lr;
                    const int h = n >> 6;                  // DH = 64
                    const int d = n & 63;
                    if (MODE == 1)
                        Yb[(((size_t)(b * NH + h)) * TSEQ + t) * DH + d] =
                            f2bf(acc[rt][nt][r]);
                    else
                        Yb[(((size_t)(b * NH + h)) * DH + d) * TSEQ + t] =
                            f2bf(acc[rt][nt][r]);
                }
            }
        }
    }
}

// ---------------------------------------------------------------------------
// Flash attention helper: one 64-key chunk (4 score tiles) consumed from the
// block-shared LDS tiles Ksh[64][72] / Vsh[64][72].
// MASKED=true only for the final diagonal chunk (j == Q0 for all waves).
// Row-sum of P is computed with a WMMA against an all-ones B matrix.
// ---------------------------------------------------------------------------
template <bool MASKED>
__device__ __forceinline__ void flash_chunk(
    const u16* __restrict__ Ksh, const u16* __restrict__ Vsh,
    u16* __restrict__ pl,                 // this wave's P staging (row stride 72)
    int jglob, int q0, int half, int lr,
    v16bf aq0, v16bf aq1, v16bf onesB,
    float* m_st, float* l_st, v8f* o) {
    const float scale = 0.125f;           // 1/sqrt(64)

    // ---- S = Q K^T : 4 tiles of 16 keys, K=64 in two k-steps ----
    v8f s[4];
#pragma unroll
    for (int ti = 0; ti < 4; ++ti) {
        const u16* krow = Ksh + (size_t)(ti * 16 + lr) * 72;
        v16bf b0 = cat16(*(const v8bf*)(krow + half * 16),
                         *(const v8bf*)(krow + half * 16 + 8));
        v16bf b1 = cat16(*(const v8bf*)(krow + 32 + half * 16),
                         *(const v8bf*)(krow + 32 + half * 16 + 8));
        v8f t = {};
        t = wmma_bf16(aq0, b0, t);
        t = wmma_bf16(aq1, b1, t);
        s[ti] = t;
    }

    // ---- online softmax state update + stage P to LDS ----
#pragma unroll
    for (int r = 0; r < 8; ++r) {
        const int row = q0 + r + half * 8;
        float a[4];
#pragma unroll
        for (int ti = 0; ti < 4; ++ti) {
            a[ti] = s[ti][r] * scale;
            if (MASKED) {
                if (jglob + ti * 16 + lr > row) a[ti] = -__builtin_inff();
            }
        }
        float mx = fmaxf(fmaxf(a[0], a[1]), fmaxf(a[2], a[3]));
        mx = fmaxf(mx, __shfl_xor(mx, 1, 32));
        mx = fmaxf(mx, __shfl_xor(mx, 2, 32));
        mx = fmaxf(mx, __shfl_xor(mx, 4, 32));
        mx = fmaxf(mx, __shfl_xor(mx, 8, 32));
        const float m_new = fmaxf(m_st[r], mx);
        const float corr = __expf(m_st[r] - m_new);
        m_st[r] = m_new;
        l_st[r] *= corr;
#pragma unroll
        for (int nt = 0; nt < 4; ++nt) o[nt][r] *= corr;
        u16* prow = pl + (size_t)(r + half * 8) * 72;
#pragma unroll
        for (int ti = 0; ti < 4; ++ti)
            prow[ti * 16 + lr] = f2bf(__expf(a[ti] - m_new));
    }
    asm volatile("s_wait_dscnt 0" ::: "memory");

    // reload P as A operand (row lr, contiguous K half per lane half)
    const u16* pr = pl + (size_t)lr * 72;
    v16bf ap0 = cat16(*(const v8bf*)(pr + half * 16),
                      *(const v8bf*)(pr + half * 16 + 8));
    v16bf ap1 = cat16(*(const v8bf*)(pr + 32 + half * 16),
                      *(const v8bf*)(pr + 32 + half * 16 + 8));

    // ---- row sums via WMMA: rs = P @ ones ----
    v8f rs = {};
    rs = wmma_bf16(ap0, onesB, rs);
    rs = wmma_bf16(ap1, onesB, rs);
#pragma unroll
    for (int r = 0; r < 8; ++r) l_st[r] += rs[r];

    // ---- O += P V (K = 64 keys, N = 64 d in 4 tiles) ----
#pragma unroll
    for (int nt = 0; nt < 4; ++nt) {
        const u16* vrow = Vsh + (size_t)(nt * 16 + lr) * 72;
        v16bf bv0 = cat16(*(const v8bf*)(vrow + half * 16),
                          *(const v8bf*)(vrow + half * 16 + 8));
        v16bf bv1 = cat16(*(const v8bf*)(vrow + 32 + half * 16),
                          *(const v8bf*)(vrow + 32 + half * 16 + 8));
        o[nt] = wmma_bf16(ap0, bv0, o[nt]);
        o[nt] = wmma_bf16(ap1, bv1, o[nt]);
    }
}

// ---------------------------------------------------------------------------
// Flash attention: block = 4 waves covering queries Q0..Q0+63 of one (b,h).
// All waves need identical K/V chunks (last chunk j==Q0 is the diagonal for
// every wave), so K/V are staged cooperatively into LDS with async copies.
// Q,K: [B,H,T,HD] bf16.  Vt: [B,H,HD,T] bf16.  ctx out: [B,T,C] bf16.
// ---------------------------------------------------------------------------
__global__ __launch_bounds__(128)
void flash_attn_kernel(const u16* __restrict__ Q, const u16* __restrict__ K,
                       const u16* __restrict__ Vt, u16* __restrict__ ctx) {
    __shared__ alignas(16) u16 Ksh[64][72];       // keys j..j+63  x d 0..63
    __shared__ alignas(16) u16 Vsh[64][72];       // d 0..63      x keys j..j+63
    __shared__ alignas(16) u16 plds[4][16][72];   // per-wave P staging

    const int tid  = threadIdx.x;
    const int wid  = tid >> 5;
    const int lane = tid & 31;
    const int half = lane >> 4;
    const int lr   = lane & 15;

    const int bh = blockIdx.y;                    // b*NH + h
    const int b  = bh >> 4;
    const int h  = bh & 15;
    const int Q0 = blockIdx.x * 64;
    const int q0 = Q0 + wid * 16;

    const u16* Qb = Q  + (size_t)bh * TSEQ * DH;
    const u16* Kb = K  + (size_t)bh * TSEQ * DH;
    const u16* Vb = Vt + (size_t)bh * DH * TSEQ;
    u16* pl = &plds[wid][0][0];

    // Q as A operand: two K-steps over d (0..31, 32..63)
    const u16* qrow = Qb + (size_t)(q0 + lr) * DH;
    v16bf aq0 = cat16(*(const v8bf*)(qrow + half * 16),
                      *(const v8bf*)(qrow + half * 16 + 8));
    v16bf aq1 = cat16(*(const v8bf*)(qrow + 32 + half * 16),
                      *(const v8bf*)(qrow + 32 + half * 16 + 8));

    v16bf onesB;
#pragma unroll
    for (int i = 0; i < 16; ++i) onesB[i] = (__bf16)1.0f;

    float m_st[8], l_st[8];
    v8f o[4] = {};
#pragma unroll
    for (int r = 0; r < 8; ++r) { m_st[r] = -__builtin_inff(); l_st[r] = 0.f; }

    for (int j = 0; j <= Q0; j += 64) {
        // ---- cooperative async staging: K chunk (8KB) + V chunk (8KB) ----
#pragma unroll
        for (int u = 0; u < 4; ++u) {
            const int unit = tid + u * 128;       // 0..511 16B units
            const int row  = unit >> 3;           // 0..63
            const int c8   = (unit & 7) * 8;      // element offset
            stage16(&Ksh[row][c8], Kb + (size_t)(j + row) * DH + c8);
            stage16(&Vsh[row][c8], Vb + (size_t)row * TSEQ + j + c8);
        }
        stage_fence();                            // s_wait_asynccnt 0 + barrier

        if (j < Q0)
            flash_chunk<false>(&Ksh[0][0], &Vsh[0][0], pl, j, q0, half, lr,
                               aq0, aq1, onesB, m_st, l_st, o);
        else
            flash_chunk<true>(&Ksh[0][0], &Vsh[0][0], pl, j, q0, half, lr,
                              aq0, aq1, onesB, m_st, l_st, o);
        __syncthreads();                          // protect LDS reuse
    }

    // normalize and store ctx (bf16, token-major [B,T,C])
#pragma unroll
    for (int r = 0; r < 8; ++r) {
        const float inv = 1.0f / l_st[r];
        const int m = q0 + r + half * 8;
        u16* crow = ctx + ((size_t)(b * TSEQ + m)) * CDIM + h * DH;
#pragma unroll
        for (int nt = 0; nt < 4; ++nt)
            crow[nt * 16 + lr] = f2bf(o[nt][r] * inv);
    }
}

// ---------------------------------------------------------------------------
// Host-side launch. Workspace layout (needs 88 MiB):
//   [0,16M)   xbf      [16M,32M) qbf     [32M,48M) kbf
//   [48M,64M) vtbf     [64M,80M) ctxbf
//   [80M..88M) wq/wk/wv/wo bf16 (2 MiB each)
// ---------------------------------------------------------------------------
extern "C" void kernel_launch(void* const* d_in, const int* in_sizes, int n_in,
                              void* d_out, int out_size, void* d_ws, size_t ws_size,
                              hipStream_t stream) {
    const float* x  = (const float*)d_in[0];
    const float* Wq = (const float*)d_in[1];
    const float* Wk = (const float*)d_in[2];
    const float* Wv = (const float*)d_in[3];
    const float* Wo = (const float*)d_in[4];
    float* out = (float*)d_out;

    char* ws = (char*)d_ws;
    const size_t MB = 1u << 20;
    u16* xbf   = (u16*)(ws + 0 * MB);
    u16* qbf   = (u16*)(ws + 16 * MB);
    u16* kbf   = (u16*)(ws + 32 * MB);
    u16* vtbf  = (u16*)(ws + 48 * MB);
    u16* ctxbf = (u16*)(ws + 64 * MB);
    u16* wqbf  = (u16*)(ws + 80 * MB);
    u16* wkbf  = (u16*)(ws + 82 * MB);
    u16* wvbf  = (u16*)(ws + 84 * MB);
    u16* wobf  = (u16*)(ws + 86 * MB);

    const int nx = MROWS * CDIM;          // 8388608
    const int nw = CDIM * CDIM;           // 1048576
    cvt_f32_bf16_kernel<<<2048, 256, 0, stream>>>(x,  xbf,  nx);
    cvt_f32_bf16_kernel<<<1024, 256, 0, stream>>>(Wq, wqbf, nw);
    cvt_f32_bf16_kernel<<<1024, 256, 0, stream>>>(Wk, wkbf, nw);
    cvt_f32_bf16_kernel<<<1024, 256, 0, stream>>>(Wv, wvbf, nw);
    cvt_f32_bf16_kernel<<<1024, 256, 0, stream>>>(Wo, wobf, nw);

    // QKV projections (WMMA GEMMs), 32x64 tile per wave
    dim3 gg(MROWS / 32, CDIM / 256);      // (256, 4)
    dim3 gb(128);
    gemm_xwT_kernel<1><<<gg, gb, 0, stream>>>(xbf, wqbf, nullptr, qbf);
    gemm_xwT_kernel<1><<<gg, gb, 0, stream>>>(xbf, wkbf, nullptr, kbf);
    gemm_xwT_kernel<2><<<gg, gb, 0, stream>>>(xbf, wvbf, nullptr, vtbf);

    // Flash attention: grid = (T/64 query groups, B*H), 4 waves per block
    flash_attn_kernel<<<dim3(TSEQ / 64, BATCH * NH), 128, 0, stream>>>(
        qbf, kbf, vtbf, ctxbf);

    // Output projection -> fp32 d_out
    gemm_xwT_kernel<0><<<gg, gb, 0, stream>>>(ctxbf, wobf, out, nullptr);
}
